// EfficientIPGLayer_67164698575280
// MI455X (gfx1250) — matrix operand order repaired and checked by reading, compiled, not run on hardware
//
#include <hip/hip_runtime.h>
#include <hip/hip_bf16.h>

// ---------------- problem constants ----------------
#define BATCH 2
#define CH    128
#define HH    48
#define WW    48
#define PIX   (HH*WW)          // 2304
#define MTOT  (BATCH*PIX)      // 4608
#define WIN   13
#define PADR  6
#define KK2   169
#define HID   256
#define NGRP  32

typedef __attribute__((ext_vector_type(16))) _Float16 v16h;
typedef __attribute__((ext_vector_type(8)))  _Float16 v8h;
typedef __attribute__((ext_vector_type(8)))  float    v8f;

union V16U { v16h v; v8h h[2]; };

__device__ __forceinline__ int reflct(int v, int n) {
    if (v < 0) v = -v;
    if (v >= n) v = 2*n - 2 - v;
    return v;
}

// ---------------- K1: transpose x [B,C,P] -> xt [B*P, C] ----------------
__global__ void k_transpose(const float* __restrict__ x, float* __restrict__ xt) {
    int idx = blockIdx.x * 256 + threadIdx.x;           // over B*P*C
    if (idx >= MTOT * CH) return;
    int pg = idx / CH;          // global pixel 0..4607
    int c  = idx % CH;
    int b  = pg / PIX;
    int p  = pg % PIX;
    xt[idx] = x[(b * CH + c) * PIX + p];
}

// ---------------- K2: bilinear 2x downsample (2x2 avg) ----------------
__global__ void k_down(const float* __restrict__ x, float* __restrict__ xd) {
    int idx = blockIdx.x * 256 + threadIdx.x;           // over B*C*24*24
    if (idx >= BATCH * CH * 24 * 24) return;
    int j  = idx % 24;
    int i  = (idx / 24) % 24;
    int bc = idx / (24 * 24);
    const float* src = x + bc * PIX;
    int y = 2 * i, xx = 2 * j;
    xd[idx] = 0.25f * (src[y*WW + xx] + src[y*WW + xx + 1] +
                       src[(y+1)*WW + xx] + src[(y+1)*WW + xx + 1]);
}

// ---------------- K3: per-pixel df (detail map) and rn (1/L2 norm) ----------------
__global__ void k_df_rn(const float* __restrict__ xt, const float* __restrict__ xd,
                        float* __restrict__ df, float* __restrict__ rn) {
    int idx = blockIdx.x * 256 + threadIdx.x;           // over B*P
    if (idx >= MTOT) return;
    int b  = idx / PIX;
    int p  = idx % PIX;
    int y  = p / WW, xx = p % WW;

    // 1 / max(||x_p||, 1e-12)
    const float4* xr = (const float4*)(xt + (size_t)idx * CH);
    float ss = 0.f;
    #pragma unroll 8
    for (int t = 0; t < CH/4; ++t) {
        float4 v = xr[t];
        ss += v.x*v.x + v.y*v.y + v.z*v.z + v.w*v.w;
    }
    rn[idx] = 1.f / fmaxf(sqrtf(ss), 1e-12f);

    // bilinear upsample sample position (jax resize, antialias=False)
    float sy = 0.5f * (float)y - 0.25f;
    float sx = 0.5f * (float)xx - 0.25f;
    int iy = (int)floorf(sy); float fy = sy - (float)iy;
    int ix = (int)floorf(sx); float fx = sx - (float)ix;
    int y0 = min(max(iy, 0), 23),  y1 = min(max(iy + 1, 0), 23);
    int x0 = min(max(ix, 0), 23),  x1 = min(max(ix + 1, 0), 23);
    float w00 = (1.f-fy)*(1.f-fx), w01 = (1.f-fy)*fx, w10 = fy*(1.f-fx), w11 = fy*fx;

    float s = 0.f;
    for (int c = 0; c < CH; ++c) {
        const float* dch = xd + ((size_t)(b*CH + c)) * 24 * 24;
        float up = w00*dch[y0*24 + x0] + w01*dch[y0*24 + x1] +
                   w10*dch[y1*24 + x0] + w11*dch[y1*24 + x1];
        s += fabsf(xt[(size_t)idx * CH + c] - up);
    }
    df[idx] = s;
}

// ---------------- K4: per-batch min/max of df ----------------
__global__ void k_minmax(const float* __restrict__ df, float* __restrict__ mm) {
    __shared__ float smin[256], smax[256];
    int b = blockIdx.x, t = threadIdx.x;
    float lmin = 3.4e38f, lmax = -3.4e38f;
    for (int i = t; i < PIX; i += 256) {
        float v = df[b * PIX + i];
        lmin = fminf(lmin, v); lmax = fmaxf(lmax, v);
    }
    smin[t] = lmin; smax[t] = lmax;
    __syncthreads();
    for (int s = 128; s > 0; s >>= 1) {
        if (t < s) { smin[t] = fminf(smin[t], smin[t+s]); smax[t] = fmaxf(smax[t], smax[t+s]); }
        __syncthreads();
    }
    if (t == 0) { mm[b*2] = smin[0]; mm[b*2+1] = smax[0]; }
}

// ---------------- K5: GroupNorm stats (mu, rstd per (b,group)) ----------------
__global__ void k_gnstats(const float* __restrict__ xt, float* __restrict__ gn) {
    __shared__ float ssum[256], ssq[256];
    int bg = blockIdx.x;                 // b*32 + g
    int b = bg / NGRP, g = bg % NGRP;
    int t = threadIdx.x;
    float s = 0.f, sq = 0.f;
    for (int p = t; p < PIX; p += 256) {
        float4 v = *(const float4*)(xt + ((size_t)(b*PIX + p)) * CH + g*4);
        s  += v.x + v.y + v.z + v.w;
        sq += v.x*v.x + v.y*v.y + v.z*v.z + v.w*v.w;
    }
    ssum[t] = s; ssq[t] = sq;
    __syncthreads();
    for (int st = 128; st > 0; st >>= 1) {
        if (t < st) { ssum[t] += ssum[t+st]; ssq[t] += ssq[t+st]; }
        __syncthreads();
    }
    if (t == 0) {
        const float inv = 1.f / (float)(PIX * (CH / NGRP));
        float mu  = ssum[0] * inv;
        float var = ssq[0] * inv - mu * mu;
        gn[bg*2]   = mu;
        gn[bg*2+1] = rsqrtf(var + 1e-5f);
    }
}

// ---------------- K6: fused local attention + GroupNorm + residual ----------------
// 8 waves/block, one pixel per wave. Emits enhanced as f32 and f16 (pixel-major).
__global__ void k_attn(const float* __restrict__ xt, const float* __restrict__ rn,
                       const float* __restrict__ df, const float* __restrict__ mm,
                       const float* __restrict__ gn,
                       const float* __restrict__ gamma, const float* __restrict__ beta,
                       float* __restrict__ Ef, _Float16* __restrict__ Eh) {
    __shared__ float xp_sh[8][CH];
    __shared__ float sim_sh[8][176];
    __shared__ float wexp_sh[8][176];

    int tid  = threadIdx.x;
    int w    = tid >> 5;
    int lane = tid & 31;
    int pg   = blockIdx.x * 8 + w;       // 0..4607
    int b    = pg / PIX;
    int p    = pg % PIX;
    int y    = p / WW, xx = p % WW;

    // load this pixel's feature vector into LDS
    #pragma unroll
    for (int t = 0; t < 4; ++t)
        xp_sh[w][lane + 32*t] = xt[(size_t)pg * CH + lane + 32*t];
    __syncthreads();

    float rnp = rn[pg];

    // --- similarity to all 169 neighbors (lane-parallel over neighbors) ---
    for (int it = 0; it < 6; ++it) {
        int k = lane + it * 32;
        if (k < KK2) {
            int di = k / WIN - PADR, dj = k % WIN - PADR;
            int qg = b * PIX + reflct(y + di, HH) * WW + reflct(xx + dj, WW);
            const float4* xq = (const float4*)(xt + (size_t)qg * CH);
            const float4* xp = (const float4*)&xp_sh[w][0];
            float dot = 0.f;
            #pragma unroll 8
            for (int t = 0; t < CH/4; ++t) {
                float4 a = xq[t]; float4 c = xp[t];
                dot += a.x*c.x + a.y*c.y + a.z*c.z + a.w*c.w;
            }
            sim_sh[w][k] = dot * rnp * rn[qg];
        }
    }
    __syncthreads();

    // --- variable top-k via stable ranks, exp weights ---
    float dmin = mm[b*2], dmax = mm[b*2+1];
    float dn  = (df[pg] - dmin) / (dmax - dmin + 1e-8f);
    int   kpi = 1 + (int)rintf(dn * 15.f);

    for (int it = 0; it < 6; ++it) {
        int k = lane + it * 32;
        if (k < KK2) {
            float si = sim_sh[w][k];
            int rank = 0;
            for (int j = 0; j < KK2; ++j) {
                float sj = sim_sh[w][j];
                rank += (sj > si) || (sj == si && j < k);
            }
            wexp_sh[w][k] = (rank < kpi) ? __expf(si) : 0.f;
        }
    }
    __syncthreads();

    float wsum = 0.f;
    for (int j = 0; j < KK2; ++j) wsum += wexp_sh[w][j];   // broadcast reads
    float winv = 1.f / wsum;

    // --- aggregate selected neighbors (weights wave-uniform -> coherent skip) ---
    float acc0 = 0.f, acc1 = 0.f, acc2 = 0.f, acc3 = 0.f;
    for (int k = 0; k < KK2; ++k) {
        float wk = wexp_sh[w][k];
        if (wk > 0.f) {
            int di = k / WIN - PADR, dj = k % WIN - PADR;
            int qg = b * PIX + reflct(y + di, HH) * WW + reflct(xx + dj, WW);
            const float* base = xt + (size_t)qg * CH;
            acc0 += wk * base[lane      ];
            acc1 += wk * base[lane + 32 ];
            acc2 += wk * base[lane + 64 ];
            acc3 += wk * base[lane + 96 ];
        }
    }

    float acc[4] = {acc0, acc1, acc2, acc3};
    #pragma unroll
    for (int t = 0; t < 4; ++t) {
        int c = lane + 32*t;
        int g = c >> 2;                          // 4 channels per group
        float mu   = gn[(b*NGRP + g)*2];
        float rstd = gn[(b*NGRP + g)*2 + 1];
        float xv = xp_sh[w][c];
        float en = acc[t] * winv + (xv - mu) * rstd * gamma[c] + beta[c];
        Ef[(size_t)pg * CH + c] = en;
        Eh[(size_t)pg * CH + c] = (_Float16)en;
    }
}

// ---------------- K7: convert FFN weights to f16 ----------------
__global__ void k_cvtw(const float* __restrict__ w1, const float* __restrict__ w2,
                       _Float16* __restrict__ w1h, _Float16* __restrict__ w2h) {
    int idx = blockIdx.x * 256 + threadIdx.x;
    if (idx >= HID * CH) return;                 // 32768 each
    w1h[idx] = (_Float16)w1[idx];
    w2h[idx] = (_Float16)w2[idx];
}

// ---------------- WMMA fragment loaders (CDNA5 16-bit layouts) ----------------
// A 16x32: lane r=lane&15 is row; hi=lane>>4 selects k-halves {hi*8..+7, 16+hi*8..+7}
__device__ __forceinline__ v16h load_fragA(const _Float16* __restrict__ A, int lda,
                                           int m0, int k0, int lane) {
    int r = lane & 15, hi = lane >> 4;
    const _Float16* base = A + (size_t)(m0 + r) * lda + k0 + hi * 8;
    V16U u;
    u.h[0] = *(const v8h*)(base);
    u.h[1] = *(const v8h*)(base + 16);
    return u.v;
}
// B 32x16 (K-major source, rows = output features): lane col = lane&15,
// hi=lane>>4 selects k-halves {hi*8..+7, 16+hi*8..+7}
__device__ __forceinline__ v16h load_fragB(const _Float16* __restrict__ B, int ldb,
                                           int n0, int k0, int lane) {
    int c = lane & 15, hi = lane >> 4;
    const _Float16* base = B + (size_t)(n0 + c) * ldb + k0 + hi * 8;
    V16U u;
    u.h[0] = *(const v8h*)(base);
    u.h[1] = *(const v8h*)(base + 16);
    return u.v;
}

// ---------------- K8: GEMM1  H = relu(E @ W1^T + b1)  [4608x128]x[128->256] ----------------
__global__ void k_gemm1(const _Float16* __restrict__ Eh, const _Float16* __restrict__ W1h,
                        const float* __restrict__ b1, _Float16* __restrict__ Hh) {
    int wave = threadIdx.x >> 5, lane = threadIdx.x & 31;
    int m0 = blockIdx.x * 16;
    int n0 = (blockIdx.y * 8 + wave) * 16;

    v8f acc = {0.f,0.f,0.f,0.f,0.f,0.f,0.f,0.f};
    #pragma unroll
    for (int k0 = 0; k0 < CH; k0 += 32) {
        v16h a = load_fragA(Eh,  CH, m0, k0, lane);
        v16h bm = load_fragB(W1h, CH, n0, k0, lane);
        acc = __builtin_amdgcn_wmma_f32_16x16x32_f16(false, a, false, bm,
                                                     (short)0, acc, false, false);
    }
    int hi = lane >> 4;
    int cg = n0 + (lane & 15);
    float bias = b1[cg];
    #pragma unroll
    for (int r = 0; r < 8; ++r) {
        int m = m0 + hi * 8 + r;
        float v = acc[r] + bias;
        Hh[(size_t)m * HID + cg] = (_Float16)fmaxf(v, 0.f);
    }
}

// ---------------- K9: GEMM2  out = E + (H @ W2^T + b2), stored as [B,C,H,W] ----------------
__global__ void k_gemm2(const _Float16* __restrict__ Hh, const _Float16* __restrict__ W2h,
                        const float* __restrict__ b2, const float* __restrict__ Ef,
                        float* __restrict__ out) {
    int wave = threadIdx.x >> 5, lane = threadIdx.x & 31;
    int m0 = blockIdx.x * 16;
    int n0 = wave * 16;

    v8f acc = {0.f,0.f,0.f,0.f,0.f,0.f,0.f,0.f};
    #pragma unroll
    for (int k0 = 0; k0 < HID; k0 += 32) {
        v16h a = load_fragA(Hh,  HID, m0, k0, lane);
        v16h bm = load_fragB(W2h, HID, n0, k0, lane);
        acc = __builtin_amdgcn_wmma_f32_16x16x32_f16(false, a, false, bm,
                                                     (short)0, acc, false, false);
    }
    int hi = lane >> 4;
    int cg = n0 + (lane & 15);
    float bias = b2[cg];
    int b  = m0 / PIX;                  // tile never crosses batch (2304 % 16 == 0)
    int p0 = m0 - b * PIX + hi * 8;

    float v[8];
    #pragma unroll
    for (int r = 0; r < 8; ++r) {
        int m = m0 + hi * 8 + r;
        v[r] = acc[r] + bias + Ef[(size_t)m * CH + cg];
    }
    float* dst = out + (size_t)(b * CH + cg) * PIX + p0;
    *(float4*)(dst)     = make_float4(v[0], v[1], v[2], v[3]);
    *(float4*)(dst + 4) = make_float4(v[4], v[5], v[6], v[7]);
}

// ---------------- host: launch pipeline ----------------
extern "C" void kernel_launch(void* const* d_in, const int* in_sizes, int n_in,
                              void* d_out, int out_size, void* d_ws, size_t ws_size,
                              hipStream_t stream) {
    const float* x     = (const float*)d_in[0];
    const float* gamma = (const float*)d_in[1];
    const float* beta  = (const float*)d_in[2];
    const float* w1    = (const float*)d_in[3];
    const float* b1    = (const float*)d_in[4];
    const float* w2    = (const float*)d_in[5];
    const float* b2    = (const float*)d_in[6];
    float* out = (float*)d_out;

    char* ws = (char*)d_ws;
    size_t off = 0;
    auto take = [&](size_t bytes) { size_t o = off; off = (off + bytes + 255) & ~(size_t)255; return o; };

    float*    xt  = (float*)   (ws + take((size_t)MTOT * CH * 4));
    float*    xd  = (float*)   (ws + take((size_t)BATCH * CH * 24 * 24 * 4));
    float*    dfv = (float*)   (ws + take((size_t)MTOT * 4));
    float*    rnv = (float*)   (ws + take((size_t)MTOT * 4));
    float*    mm  = (float*)   (ws + take((size_t)BATCH * 2 * 4));
    float*    gnv = (float*)   (ws + take((size_t)BATCH * NGRP * 2 * 4));
    float*    Ef  = (float*)   (ws + take((size_t)MTOT * CH * 4));
    _Float16* Eh  = (_Float16*)(ws + take((size_t)MTOT * CH * 2));
    _Float16* W1h = (_Float16*)(ws + take((size_t)HID * CH * 2));
    _Float16* W2h = (_Float16*)(ws + take((size_t)HID * CH * 2));
    _Float16* Hh  = (_Float16*)(ws + take((size_t)MTOT * HID * 2));
    (void)ws_size; (void)n_in; (void)in_sizes; (void)out_size;

    k_transpose<<<(MTOT * CH + 255) / 256, 256, 0, stream>>>(x, xt);
    k_down<<<(BATCH * CH * 24 * 24 + 255) / 256, 256, 0, stream>>>(x, xd);
    k_df_rn<<<(MTOT + 255) / 256, 256, 0, stream>>>(xt, xd, dfv, rnv);
    k_minmax<<<BATCH, 256, 0, stream>>>(dfv, mm);
    k_gnstats<<<BATCH * NGRP, 256, 0, stream>>>(xt, gnv);
    k_attn<<<MTOT / 8, 256, 0, stream>>>(xt, rnv, dfv, mm, gnv, gamma, beta, Ef, Eh);
    k_cvtw<<<(HID * CH + 255) / 256, 256, 0, stream>>>(w1, w2, W1h, W2h);
    // GEMM1: M=4608 (288 tiles), N=256 (16 tiles -> grid.y=2, 8 waves each)
    k_gemm1<<<dim3(MTOT / 16, 2), 256, 0, stream>>>(Eh, W1h, b1, Hh);
    // GEMM2: M=4608, N=128 (8 tiles -> 8 waves)
    k_gemm2<<<dim3(MTOT / 16, 1), 256, 0, stream>>>(Hh, W2h, b2, Ef, out);
}
